// MultiHeadAttention_28733331211025
// MI455X (gfx1250) — matrix operand order repaired and checked by reading, not compiled, not run on hardware
//
#include <hip/hip_runtime.h>
#include <hip/hip_bf16.h>
#include <math.h>

// ---------------------------------------------------------------------------
// MHA forward for MI455X (gfx1250, wave32, WMMA).
// B=4, S=2048, D=1024, H=16, HD=64.
// out (B,S,D) f32  ++  attn_weights (B,H,S,S) f32
// ---------------------------------------------------------------------------

typedef __bf16 bf16_t;
typedef __attribute__((ext_vector_type(16))) __bf16 v16bf;
typedef __attribute__((ext_vector_type(8)))  __bf16 v8bf;
typedef __attribute__((ext_vector_type(8)))  float  v8f;
typedef __attribute__((ext_vector_type(4)))  float  v4f;
typedef __attribute__((ext_vector_type(4)))  int    v4i;

#define GLOBAL_AS __attribute__((address_space(1)))
#define LDS_AS    __attribute__((address_space(3)))

static constexpr int Bc  = 4;
static constexpr int Sc  = 2048;
static constexpr int Dc  = 1024;
static constexpr int Hc  = 16;
static constexpr int HDc = 64;
static constexpr int Mc  = Bc * Sc;   // 8192

// LDS B-tile (transposed layout): N-major [n][k], row stride 40 (80B).
static constexpr int LDS_STRIDE = 40;

#if defined(__HIP_DEVICE_COMPILE__) && \
    __has_builtin(__builtin_amdgcn_global_load_async_to_lds_b128) && \
    __has_builtin(__builtin_amdgcn_s_wait_asynccnt)
#define USE_ASYNC_LDS 1
#else
#define USE_ASYNC_LDS 0
#endif

// ---------------------------------------------------------------------------
// f32 -> bf16 conversion (x staging)
// ---------------------------------------------------------------------------
__global__ __launch_bounds__(256) void cvt_f32_bf16_kernel(const float* __restrict__ src,
                                                           bf16_t* __restrict__ dst,
                                                           int n) {
  int base = (blockIdx.x * 256 + threadIdx.x) * 4;
  if (base + 3 < n) {
    v4f v = *(const v4f*)(src + base);
    dst[base + 0] = (bf16_t)v.x;
    dst[base + 1] = (bf16_t)v.y;
    dst[base + 2] = (bf16_t)v.z;
    dst[base + 3] = (bf16_t)v.w;
  }
}

// ---------------------------------------------------------------------------
// GEMM: C[M,N] = A_bf16[M,K] * W_f32[K,N] + bias_f32[N]
// Block = 256 thr = 8 waves, macro-tile 128(M) x 64(N); wave = 64x16 via 4 acc.
// B tile staged transposed in LDS per 32-wide K-step (f32->bf16 in flight);
// fragments read with two ds_load_b128 per lane.
// ---------------------------------------------------------------------------
template <typename OutT>
__global__ __launch_bounds__(256) void gemm_bf16_kernel(const bf16_t* __restrict__ A,
                                                        const float* __restrict__ W,
                                                        const float* __restrict__ bias,
                                                        OutT* __restrict__ C,
                                                        int M, int N, int K) {
  __shared__ bf16_t ldsB[64][LDS_STRIDE];

  const int tid    = threadIdx.x;
  const int lane   = tid & 31;
  const int wv     = tid >> 5;          // 0..7
  const int mGroup = wv >> 2;           // 0..1
  const int nGroup = wv & 3;            // 0..3

  const int nBlocks = N >> 6;
  const int blockM  = blockIdx.x / nBlocks;
  const int blockN  = blockIdx.x % nBlocks;

  const int mBase  = blockM * 128 + mGroup * 64;
  const int half   = lane >> 4;
  const int l16    = lane & 15;
  const int nLocal = nGroup * 16 + l16;
  const int ncol   = blockN * 64 + nLocal;

  const int ck  = tid >> 3;             // 0..31
  const int cn0 = (tid & 7) * 8;        // 0,8,..,56

  v8f acc[4] = {};

  for (int kk = 0; kk < K; kk += 32) {
    __syncthreads();
    // ---- stage B tile (f32 -> bf16, transposed to N-major) ----
    {
      const float* wr = W + (size_t)(kk + ck) * N + blockN * 64 + cn0;
      v4f w0 = *(const v4f*)wr;
      v4f w1 = *(const v4f*)(wr + 4);
#pragma unroll
      for (int j = 0; j < 4; ++j) ldsB[cn0 + j][ck]     = (bf16_t)w0[j];
#pragma unroll
      for (int j = 0; j < 4; ++j) ldsB[cn0 + 4 + j][ck] = (bf16_t)w1[j];
    }
    __syncthreads();

    const int kb = half * 16;
    v8bf b0 = *(const v8bf*)&ldsB[nLocal][kb];
    v8bf b1 = *(const v8bf*)&ldsB[nLocal][kb + 8];
    v16bf bfv;
#pragma unroll
    for (int i = 0; i < 8; ++i) { bfv[i] = b0[i]; bfv[i + 8] = b1[i]; }

#pragma unroll
    for (int sub = 0; sub < 4; ++sub) {
      const bf16_t* ar = A + (size_t)(mBase + sub * 16 + l16) * K + kk + half * 16;
      __builtin_prefetch(ar + 32, 0, 1);   // next K-step (global_prefetch_b8)
      v8bf a0 = *(const v8bf*)ar;
      v8bf a1 = *(const v8bf*)(ar + 8);
      v16bf af;
#pragma unroll
      for (int i = 0; i < 8; ++i) { af[i] = a0[i]; af[i + 8] = a1[i]; }
      acc[sub] = __builtin_amdgcn_wmma_f32_16x16x32_bf16(false, af, false, bfv,
                                                         (short)0, acc[sub], false, false);
    }
  }

  const float bv = bias[ncol];
#pragma unroll
  for (int sub = 0; sub < 4; ++sub) {
#pragma unroll
    for (int r = 0; r < 8; ++r) {
      const int row = mBase + sub * 16 + r + half * 8;
      C[(size_t)row * N + ncol] = (OutT)(acc[sub][r] + bv);
    }
  }
}

// ---------------------------------------------------------------------------
// Scores: SW[b,h,q,kv] = (Q_h K_h^T)/8 + mask[b,kv]*-1e9   (raw, pre-softmax)
// Wave = 64(q) x 16(kv) via 4 accumulators; K-dim = HD = 64 (2 steps, unrolled).
// ---------------------------------------------------------------------------
__global__ __launch_bounds__(256) void scores_kernel(const bf16_t* __restrict__ Q,
                                                     const bf16_t* __restrict__ Kc,
                                                     const float* __restrict__ mask,
                                                     float* __restrict__ SW) {
  const int wave = blockIdx.x * (blockDim.x >> 5) + (threadIdx.x >> 5);
  const int lane = threadIdx.x & 31;
  const int bh  = wave >> 12;
  const int rem = wave & 4095;
  const int qg  = rem >> 7;             // 0..31
  const int kt  = rem & 127;            // 0..127
  const int b = bh >> 4;
  const int h = bh & 15;

  const int l16  = lane & 15;
  const int half = lane >> 4;
  const int qBase = qg * 64;

  v8f acc[4] = {};
#pragma unroll
  for (int dd = 0; dd < HDc; dd += 32) {
    const bf16_t* kr = Kc + (size_t)(b * Sc + kt * 16 + l16) * Dc + h * HDc + dd + half * 16;
    v8bf b0 = *(const v8bf*)kr;
    v8bf b1 = *(const v8bf*)(kr + 8);
    v16bf bfv;
#pragma unroll
    for (int i = 0; i < 8; ++i) { bfv[i] = b0[i]; bfv[i + 8] = b1[i]; }

#pragma unroll
    for (int sub = 0; sub < 4; ++sub) {
      const bf16_t* qr = Q + (size_t)(b * Sc + qBase + sub * 16 + l16) * Dc
                           + h * HDc + dd + half * 16;
      v8bf a0 = *(const v8bf*)qr;
      v8bf a1 = *(const v8bf*)(qr + 8);
      v16bf af;
#pragma unroll
      for (int i = 0; i < 8; ++i) { af[i] = a0[i]; af[i + 8] = a1[i]; }
      acc[sub] = __builtin_amdgcn_wmma_f32_16x16x32_bf16(false, af, false, bfv,
                                                         (short)0, acc[sub], false, false);
    }
  }

  const int kv = kt * 16 + l16;
  const float madd  = mask[b * Sc + kv] * -1e9f;
  const float scale = 0.125f;           // 1/sqrt(64)
#pragma unroll
  for (int sub = 0; sub < 4; ++sub) {
#pragma unroll
    for (int r = 0; r < 8; ++r) {
      const int q = qBase + sub * 16 + r + half * 8;
      SW[(((size_t)(b * Hc + h)) * Sc + q) * Sc + kv] = acc[sub][r] * scale + madd;
    }
  }
}

// ---------------------------------------------------------------------------
// Softmax in place over last dim (S=2048). One wave per row; lane owns a
// contiguous 64-float chunk -> 16 b128 loads + 16 b128 stores per lane.
// ---------------------------------------------------------------------------
__global__ __launch_bounds__(256) void softmax_kernel(float* __restrict__ SW) {
  const int row  = blockIdx.x * (blockDim.x >> 5) + (threadIdx.x >> 5);
  const int lane = threadIdx.x & 31;
  float* r = SW + (size_t)row * Sc + lane * 64;

  v4f c[16];
#pragma unroll
  for (int i = 0; i < 16; ++i) c[i] = *(const v4f*)(r + i * 4);

  float m = -INFINITY;
#pragma unroll
  for (int i = 0; i < 16; ++i)
#pragma unroll
    for (int j = 0; j < 4; ++j) m = fmaxf(m, c[i][j]);
#pragma unroll
  for (int off = 16; off > 0; off >>= 1) m = fmaxf(m, __shfl_xor(m, off, 32));

  float s = 0.f;
#pragma unroll
  for (int i = 0; i < 16; ++i)
#pragma unroll
    for (int j = 0; j < 4; ++j) { c[i][j] = __expf(c[i][j] - m); s += c[i][j]; }
#pragma unroll
  for (int off = 16; off > 0; off >>= 1) s += __shfl_xor(s, off, 32);

  const float inv = 1.f / s;
#pragma unroll
  for (int i = 0; i < 16; ++i) {
#pragma unroll
    for (int j = 0; j < 4; ++j) c[i][j] *= inv;
    *(v4f*)(r + i * 4) = c[i];
  }
}

// ---------------------------------------------------------------------------
// ctx[b, q, h*64+n] = sum_kv SW[b,h,q,kv] * V[b,kv,h*64+n]
// Block = 8 waves over 128(q) x 64(n) for one (b,h).
// V tile staged to LDS per 32-wide K-step:
//   - async path: GLOBAL_LOAD_ASYNC_TO_LDS_B128 (row-major, ASYNCcnt) +
//     conflict-free per-element gathers for the B fragment.
//   - fallback:   transposed store + two ds_load_b128 per lane.
// ---------------------------------------------------------------------------
__global__ __launch_bounds__(256) void ctx_kernel(const float* __restrict__ SW,
                                                  const bf16_t* __restrict__ V,
                                                  bf16_t* __restrict__ CTX) {
#if USE_ASYNC_LDS
  __shared__ bf16_t ldsV[32][80];       // row-major [kv][n], 160B row stride
#else
  __shared__ bf16_t ldsB[64][LDS_STRIDE];
#endif

  const int tid    = threadIdx.x;
  const int lane   = tid & 31;
  const int wv     = tid >> 5;
  const int mGroup = wv >> 2;           // 0..1
  const int nGroup = wv & 3;            // 0..3

  const int bh = blockIdx.x >> 4;
  const int qb = blockIdx.x & 15;
  const int b = bh >> 4;
  const int h = bh & 15;

  const int l16   = lane & 15;
  const int half  = lane >> 4;
  const int qBase = qb * 128 + mGroup * 64;
  const int nLoc  = nGroup * 16 + l16;

  const int ck  = tid >> 3;             // 0..31
  const int cn0 = (tid & 7) * 8;        // 0..56

  v8f acc[4] = {};

  for (int kk = 0; kk < Sc; kk += 32) {
    __syncthreads();
    // ---- stage V tile 32(kv) x 64(n) ----
    {
      const bf16_t* vr = V + (size_t)(b * Sc + kk + ck) * Dc + h * HDc + cn0;
#if USE_ASYNC_LDS
      __builtin_amdgcn_global_load_async_to_lds_b128(
          (GLOBAL_AS v4i*)vr,
          (LDS_AS v4i*)&ldsV[ck][cn0],
          0, 0);
#else
      v8bf vv = *(const v8bf*)vr;
#pragma unroll
      for (int j = 0; j < 8; ++j) ldsB[cn0 + j][ck] = vv[j];
#endif
    }
#if USE_ASYNC_LDS
    __builtin_amdgcn_s_wait_asynccnt(0);
#endif
    __syncthreads();

    // ---- B fragment: lane holds column nLoc, 16 consecutive kv rows ----
    const int kb = half * 16;
    v16bf bfv;
#if USE_ASYNC_LDS
#pragma unroll
    for (int e = 0; e < 16; ++e) bfv[e] = ldsV[kb + e][nLoc];
#else
    {
      v8bf b0 = *(const v8bf*)&ldsB[nLoc][kb];
      v8bf b1 = *(const v8bf*)&ldsB[nLoc][kb + 8];
#pragma unroll
      for (int i = 0; i < 8; ++i) { bfv[i] = b0[i]; bfv[i + 8] = b1[i]; }
    }
#endif

#pragma unroll
    for (int sub = 0; sub < 4; ++sub) {
      const float* ar = SW + (((size_t)(b * Hc + h)) * Sc + qBase + sub * 16 + l16) * Sc
                           + kk + half * 16;
      __builtin_prefetch(ar + 32, 0, 1);
      v4f a00 = *(const v4f*)ar;
      v4f a01 = *(const v4f*)(ar + 4);
      v4f a10 = *(const v4f*)(ar + 8);
      v4f a11 = *(const v4f*)(ar + 12);
      v16bf af;
#pragma unroll
      for (int i = 0; i < 4; ++i) {
        af[i]      = (bf16_t)a00[i];
        af[i + 4]  = (bf16_t)a01[i];
        af[i + 8]  = (bf16_t)a10[i];
        af[i + 12] = (bf16_t)a11[i];
      }
      acc[sub] = __builtin_amdgcn_wmma_f32_16x16x32_bf16(false, af, false, bfv,
                                                         (short)0, acc[sub], false, false);
    }
  }

#pragma unroll
  for (int sub = 0; sub < 4; ++sub) {
#pragma unroll
    for (int r = 0; r < 8; ++r) {
      const int q = qBase + sub * 16 + r + half * 8;
      CTX[(size_t)(b * Sc + q) * Dc + h * HDc + nLoc] = (bf16_t)acc[sub][r];
    }
  }
}

// ---------------------------------------------------------------------------
// Host-side launcher
// ---------------------------------------------------------------------------
extern "C" void kernel_launch(void* const* d_in, const int* in_sizes, int n_in,
                              void* d_out, int out_size, void* d_ws, size_t ws_size,
                              hipStream_t stream) {
  (void)in_sizes; (void)n_in; (void)out_size; (void)ws_size;

  const float* x    = (const float*)d_in[0];
  const float* mask = (const float*)d_in[1];
  const float* wq   = (const float*)d_in[2];
  const float* bq   = (const float*)d_in[3];
  const float* wk   = (const float*)d_in[4];
  const float* bk   = (const float*)d_in[5];
  const float* wv   = (const float*)d_in[6];
  const float* bv   = (const float*)d_in[7];
  const float* wo   = (const float*)d_in[8];
  const float* bo   = (const float*)d_in[9];

  float* out   = (float*)d_out;                       // (B,S,D)
  float* attnw = (float*)d_out + (size_t)Mc * Dc;     // (B,H,S,S)

  const size_t buf = (size_t)Mc * Dc;                 // elements
  bf16_t* xb  = (bf16_t*)d_ws;
  bf16_t* Q   = xb + buf;
  bf16_t* K   = Q + buf;
  bf16_t* V   = K + buf;
  bf16_t* CTX = V + buf;

  const int nElem = Mc * Dc;  // 8,388,608

  // 1) stage x as bf16
  cvt_f32_bf16_kernel<<<nElem / (256 * 4), 256, 0, stream>>>(x, xb, nElem);

  // 2) QKV projections: (8192/128)*(1024/64) = 1024 blocks
  const int gemmBlocks = (Mc / 128) * (Dc / 64);
  gemm_bf16_kernel<bf16_t><<<gemmBlocks, 256, 0, stream>>>(xb, wq, bq, Q, Mc, Dc, Dc);
  gemm_bf16_kernel<bf16_t><<<gemmBlocks, 256, 0, stream>>>(xb, wk, bk, K, Mc, Dc, Dc);
  gemm_bf16_kernel<bf16_t><<<gemmBlocks, 256, 0, stream>>>(xb, wv, bv, V, Mc, Dc, Dc);

  // 3) raw scores: B*H*(S/64)*(S/16) waves / 8 per block = 32768 blocks
  const int scoreBlocks = Bc * Hc * (Sc / 64) * (Sc / 16) / 8;
  scores_kernel<<<scoreBlocks, 256, 0, stream>>>(Q, K, mask, attnw);

  // 4) softmax in place: B*H*S rows / 8 per block
  const int smBlocks = Bc * Hc * Sc / 8;  // 16384
  softmax_kernel<<<smBlocks, 256, 0, stream>>>(attnw);

  // 5) ctx = attnw @ V: B*H * 16 q-blocks = 1024 blocks
  const int ctxBlocks = Bc * Hc * (Sc / 128);
  ctx_kernel<<<ctxBlocks, 256, 0, stream>>>(attnw, V, CTX);

  // 6) output projection -> f32 out
  gemm_bf16_kernel<float><<<gemmBlocks, 256, 0, stream>>>(CTX, wo, bo, out, Mc, Dc, Dc);
}
